// GAT_50603304681766
// MI455X (gfx1250) — matrix operand order repaired and checked by reading, not compiled, not run on hardware
//
#include <hip/hip_runtime.h>
#include <hip/hip_bf16.h>
#include <math.h>

// Problem constants (match reference)
#define N_NODES 50000
#define E_EDGES 800000
#define F_IN    256
#define HEADS   4
#define C1      64
#define C2      32
#define HC1     (HEADS * C1)   // 256
#define HC2     (HEADS * C2)   // 128
#define E_TOT   (E_EDGES + N_NODES)  // 850000 with self loops
#define NEG_SLOPE 0.2f

typedef __attribute__((ext_vector_type(16))) __bf16 v16bf;
typedef __attribute__((ext_vector_type(8)))  float  v8f;

union Frag16 { v16bf v; uint4 q[2]; };

static __device__ __forceinline__ unsigned short f2bf(float f) {
  union { float f; unsigned int u; } x; x.f = f;
  unsigned int r = x.u + 0x7FFFu + ((x.u >> 16) & 1u);   // RNE
  if ((x.u & 0x7F800000u) == 0x7F800000u) r = x.u;        // inf/nan passthrough
  return (unsigned short)(r >> 16);
}

__global__ void fill_f32(float* p, float v, int n) {
  int i = blockIdx.x * blockDim.x + threadIdx.x;
  if (i < n) p[i] = v;
}

__global__ void cast_bf16(const float* __restrict__ in, unsigned short* __restrict__ out, int n) {
  int i = blockIdx.x * blockDim.x + threadIdx.x;
  if (i < n) out[i] = f2bf(in[i]);
}

// Wt[n*K + k] = bf16(W[k*Nout + n])  (W is K x Nout row-major)
__global__ void transpose_cast(const float* __restrict__ W, unsigned short* __restrict__ Wt,
                               int K, int Nout) {
  int i = blockIdx.x * blockDim.x + threadIdx.x;
  if (i >= K * Nout) return;
  int k = i % K, n = i / K;
  Wt[i] = f2bf(W[(size_t)k * Nout + n]);
}

// C[M x Nout] = A[M x K](bf16, row major) * Bt[Nout x K](bf16, row major == B col-major)
// 256 threads = 8 waves; wave owns 16 rows x 64 cols; K stepped by 32.
__global__ void __launch_bounds__(256)
gemm_bf16_wmma(const unsigned short* __restrict__ A,
               const unsigned short* __restrict__ Bt,
               float* __restrict__ C, int M, int K, int Nout) {
  const int tid  = threadIdx.x;
  const int wave = tid >> 5;
  const int lane = tid & 31;
  const int hi   = lane >> 4;   // K-half select (A/B 16-bit fragment layout)
  const int l16  = lane & 15;
  const int m0   = blockIdx.x * 128 + wave * 16;
  const int n0   = blockIdx.y * 64;

  int arow = m0 + l16; if (arow >= M) arow = M - 1;          // clamp; store guarded
  const unsigned short* aptr = A + (size_t)arow * K + hi * 8;
  const unsigned short* bptr[4];
#pragma unroll
  for (int t = 0; t < 4; ++t)
    bptr[t] = Bt + (size_t)(n0 + t * 16 + l16) * K + hi * 8;

  v8f acc[4] = {};
  for (int kk = 0; kk < K; kk += 32) {
    Frag16 a;
    a.q[0] = *(const uint4*)(aptr + kk);        // K = kbase .. kbase+7
    a.q[1] = *(const uint4*)(aptr + kk + 16);   // K = kbase+16 .. kbase+23
    Frag16 b[4];
#pragma unroll
    for (int t = 0; t < 4; ++t) {
      b[t].q[0] = *(const uint4*)(bptr[t] + kk);      // K = kbase .. kbase+7
      b[t].q[1] = *(const uint4*)(bptr[t] + kk + 16); // K = kbase+16 .. kbase+23
    }
#pragma unroll
    for (int t = 0; t < 4; ++t)
      acc[t] = __builtin_amdgcn_wmma_f32_16x16x32_bf16(
          false, a.v, false, b[t].v, (short)0, acc[t], false, false);
  }

#pragma unroll
  for (int t = 0; t < 4; ++t) {
#pragma unroll
    for (int v = 0; v < 8; ++v) {
      int r = m0 + v + hi * 8;               // D layout: lanes>=16 hold M=v+8
      if (r < M) C[(size_t)r * Nout + n0 + t * 16 + l16] = acc[t][v];
    }
  }
}

// e_src[n,h] = sum_c h[n,h,c]*a_src[h,c]; likewise e_dst
__global__ void att_scores(const float* __restrict__ h, const float* __restrict__ a_src,
                           const float* __restrict__ a_dst, float* __restrict__ es,
                           float* __restrict__ ed, int n_nodes, int C) {
  int i = blockIdx.x * blockDim.x + threadIdx.x;
  if (i >= n_nodes * HEADS) return;
  int n = i / HEADS, hh = i % HEADS;
  const float* hp = h + (size_t)n * HEADS * C + hh * C;
  const float* as = a_src + hh * C;
  const float* ad = a_dst + hh * C;
  float ss = 0.f, sd = 0.f;
  for (int c = 0; c < C; ++c) { float v = hp[c]; ss += v * as[c]; sd += v * ad[c]; }
  es[i] = ss; ed[i] = sd;
}

static __device__ __forceinline__ int esrc(const int* ei, int e) {
  return e < E_EDGES ? ei[e] : e - E_EDGES;
}
static __device__ __forceinline__ int edst(const int* ei, int e) {
  return e < E_EDGES ? ei[E_EDGES + e] : e - E_EDGES;
}
static __device__ __forceinline__ float lrelu(float v) {
  return v > 0.f ? v : NEG_SLOPE * v;
}

__global__ void edge_max(const int* __restrict__ ei, const float* __restrict__ es,
                         const float* __restrict__ ed, float* __restrict__ m) {
  int i = blockIdx.x * blockDim.x + threadIdx.x;
  if (i >= E_TOT * HEADS) return;
  int e = i / HEADS, hh = i % HEADS;
  int s = esrc(ei, e), d = edst(ei, e);
  float v = lrelu(es[s * HEADS + hh] + ed[d * HEADS + hh]);
  float* addr = m + d * HEADS + hh;
  if (v >= 0.f) atomicMax((int*)addr, __float_as_int(v));
  else          atomicMin((unsigned int*)addr, __float_as_uint(v));
}

__global__ void edge_exp(const int* __restrict__ ei, const float* __restrict__ es,
                         const float* __restrict__ ed, const float* __restrict__ m,
                         float* __restrict__ alpha, float* __restrict__ denom) {
  int i = blockIdx.x * blockDim.x + threadIdx.x;
  if (i >= E_TOT * HEADS) return;
  int e = i / HEADS, hh = i % HEADS;
  int s = esrc(ei, e), d = edst(ei, e);
  float mm = m[d * HEADS + hh];
  if (!__builtin_isfinite(mm)) mm = 0.f;
  float v  = lrelu(es[s * HEADS + hh] + ed[d * HEADS + hh]);
  float ex = __expf(v - mm);
  alpha[i] = ex;
  atomicAdd(denom + d * HEADS + hh, ex);
}

// 32 lanes per (edge, head) pair; lanes stride channels
__global__ void edge_agg(const int* __restrict__ ei, const float* __restrict__ alpha,
                         const float* __restrict__ denom, const float* __restrict__ hsrc,
                         float* __restrict__ out, int C) {
  int pair = blockIdx.x * (blockDim.x >> 5) + (threadIdx.x >> 5);
  int lane = threadIdx.x & 31;
  if (pair >= E_TOT * HEADS) return;
  int e = pair / HEADS, hh = pair % HEADS;
  int s = esrc(ei, e), d = edst(ei, e);
  float coef = alpha[pair] / (denom[d * HEADS + hh] + 1e-16f);
  const float* hp = hsrc + (size_t)s * HEADS * C + hh * C;
  float* op = out + (size_t)d * HEADS * C + hh * C;
  for (int c = lane; c < C; c += 32)
    atomicAdd(op + c, hp[c] * coef);
}

__global__ void elu_bias_cast(const float* __restrict__ in, const float* __restrict__ b,
                              unsigned short* __restrict__ out, int n_nodes, int F) {
  int i = blockIdx.x * blockDim.x + threadIdx.x;
  if (i >= n_nodes * F) return;
  float v = in[i] + b[i % F];
  v = v > 0.f ? v : (__expf(v) - 1.f);   // ELU
  out[i] = f2bf(v);
}

// one wave per node; F == 128 (4 values per lane)
__global__ void logsoftmax_bias(const float* __restrict__ in, const float* __restrict__ b,
                                float* __restrict__ out, int n_nodes, int F) {
  int node = blockIdx.x * (blockDim.x >> 5) + (threadIdx.x >> 5);
  int lane = threadIdx.x & 31;
  if (node >= n_nodes) return;
  const float* ip = in + (size_t)node * F;
  float vals[4];
  float mx = -INFINITY;
#pragma unroll
  for (int j = 0; j < 4; ++j) {
    vals[j] = ip[lane + 32 * j] + b[lane + 32 * j];
    mx = fmaxf(mx, vals[j]);
  }
  for (int o = 16; o > 0; o >>= 1) mx = fmaxf(mx, __shfl_xor(mx, o, 32));
  float s = 0.f;
#pragma unroll
  for (int j = 0; j < 4; ++j) s += __expf(vals[j] - mx);
  for (int o = 16; o > 0; o >>= 1) s += __shfl_xor(s, o, 32);
  float lse = mx + __logf(s);
  float* op = out + (size_t)node * F;
#pragma unroll
  for (int j = 0; j < 4; ++j) op[lane + 32 * j] = vals[j] - lse;
}

extern "C" void kernel_launch(void* const* d_in, const int* in_sizes, int n_in,
                              void* d_out, int out_size, void* d_ws, size_t ws_size,
                              hipStream_t stream) {
  (void)in_sizes; (void)n_in; (void)out_size; (void)ws_size;
  const float* x     = (const float*)d_in[0];
  const int*   ei    = (const int*)  d_in[1];
  const float* W1    = (const float*)d_in[2];
  const float* asrc1 = (const float*)d_in[3];
  const float* adst1 = (const float*)d_in[4];
  const float* b1    = (const float*)d_in[5];
  const float* W2    = (const float*)d_in[6];
  const float* asrc2 = (const float*)d_in[7];
  const float* adst2 = (const float*)d_in[8];
  const float* b2    = (const float*)d_in[9];
  float* out = (float*)d_out;
  char*  ws  = (char*)d_ws;

  // Workspace layout (bytes), 256-aligned; regions reused across phases.
  const size_t o_xb    = 0;                                   // bf16 x  -> later bf16 ELU(h1)
  const size_t o_w1t   = o_xb  + (size_t)N_NODES * F_IN * 2;  // 25.6 MB
  const size_t o_w2t   = o_w1t + (size_t)F_IN * HC1 * 2;      // 128 KB
  const size_t o_h1    = o_w2t + (size_t)HC1 * HC2 * 2;       // 64 KB
  const size_t o_h2    = o_h1;                                 // reuse after layer1 agg
  const size_t o_o2    = o_h1  + (size_t)N_NODES * HC1 * 2;   // second half of h1 region
  const size_t o_es    = o_h1  + (size_t)N_NODES * HC1 * 4;   // 51.2 MB region ends
  const size_t o_ed    = o_es  + (size_t)N_NODES * HEADS * 4;
  const size_t o_m     = o_ed  + (size_t)N_NODES * HEADS * 4;
  const size_t o_dn    = o_m   + (size_t)N_NODES * HEADS * 4;
  const size_t o_alpha = o_dn  + (size_t)N_NODES * HEADS * 4;
  const size_t o_o1    = o_alpha + (size_t)E_TOT * HEADS * 4;

  unsigned short* xb   = (unsigned short*)(ws + o_xb);
  unsigned short* h1b  = (unsigned short*)(ws + o_xb);   // alias, used after xb is dead
  unsigned short* w1t  = (unsigned short*)(ws + o_w1t);
  unsigned short* w2t  = (unsigned short*)(ws + o_w2t);
  float* h1    = (float*)(ws + o_h1);
  float* h2    = (float*)(ws + o_h2);
  float* o2    = (float*)(ws + o_o2);
  float* es    = (float*)(ws + o_es);
  float* ed    = (float*)(ws + o_ed);
  float* m     = (float*)(ws + o_m);
  float* dn    = (float*)(ws + o_dn);
  float* alpha = (float*)(ws + o_alpha);
  float* o1    = (float*)(ws + o_o1);

  const int NH  = N_NODES * HEADS;
  const int EH  = E_TOT * HEADS;
  auto cdiv = [](int a, int b) { return (a + b - 1) / b; };

  // ---- casts / transposes ----
  cast_bf16<<<cdiv(N_NODES * F_IN, 256), 256, 0, stream>>>(x, xb, N_NODES * F_IN);
  transpose_cast<<<cdiv(F_IN * HC1, 256), 256, 0, stream>>>(W1, w1t, F_IN, HC1);
  transpose_cast<<<cdiv(HC1 * HC2, 256), 256, 0, stream>>>(W2, w2t, HC1, HC2);

  // ---- layer 1 ----
  fill_f32<<<cdiv(NH, 256), 256, 0, stream>>>(m,  -INFINITY, NH);
  fill_f32<<<cdiv(NH, 256), 256, 0, stream>>>(dn, 0.f, NH);
  fill_f32<<<cdiv(N_NODES * HC1, 256), 256, 0, stream>>>(o1, 0.f, N_NODES * HC1);

  gemm_bf16_wmma<<<dim3(cdiv(N_NODES, 128), HC1 / 64), 256, 0, stream>>>(
      xb, w1t, h1, N_NODES, F_IN, HC1);
  att_scores<<<cdiv(NH, 256), 256, 0, stream>>>(h1, asrc1, adst1, es, ed, N_NODES, C1);
  edge_max<<<cdiv(EH, 256), 256, 0, stream>>>(ei, es, ed, m);
  edge_exp<<<cdiv(EH, 256), 256, 0, stream>>>(ei, es, ed, m, alpha, dn);
  edge_agg<<<cdiv(EH, 8), 256, 0, stream>>>(ei, alpha, dn, h1, o1, C1);
  elu_bias_cast<<<cdiv(N_NODES * HC1, 256), 256, 0, stream>>>(o1, b1, h1b, N_NODES, HC1);

  // ---- layer 2 ----
  fill_f32<<<cdiv(NH, 256), 256, 0, stream>>>(m,  -INFINITY, NH);
  fill_f32<<<cdiv(NH, 256), 256, 0, stream>>>(dn, 0.f, NH);
  fill_f32<<<cdiv(N_NODES * HC2, 256), 256, 0, stream>>>(o2, 0.f, N_NODES * HC2);

  gemm_bf16_wmma<<<dim3(cdiv(N_NODES, 128), HC2 / 64), 256, 0, stream>>>(
      h1b, w2t, h2, N_NODES, HC1, HC2);
  att_scores<<<cdiv(NH, 256), 256, 0, stream>>>(h2, asrc2, adst2, es, ed, N_NODES, C2);
  edge_max<<<cdiv(EH, 256), 256, 0, stream>>>(ei, es, ed, m);
  edge_exp<<<cdiv(EH, 256), 256, 0, stream>>>(ei, es, ed, m, alpha, dn);
  edge_agg<<<cdiv(EH, 8), 256, 0, stream>>>(ei, alpha, dn, h2, o2, C2);

  logsoftmax_bias<<<cdiv(N_NODES, 8), 256, 0, stream>>>(o2, b2, out, N_NODES, HC2);
}
